// RES_GVP_953482740113
// MI455X (gfx1250) — matrix operand order repaired and checked by reading, not compiled, not run on hardware
//
#include <hip/hip_runtime.h>
#include <math.h>

// ============================================================================
// GVP-GNN forward for MI455X (gfx1250, wave32, WMMA f16 16x16x32).
// One workgroup (128 thr = 4 waves) handles a 16-edge or 16-node tile; every
// GEMM-shaped contraction in each GVP runs on v_wmma_f32_16x16x32_f16 with
// f32 accumulation. Weights are pre-swizzled into WMMA B-fragment order
// (one contiguous 32B vector per lane per 32x16 tile) and stay hot in L2.
// A-fragments are two ds_load_b128 per lane from 16B-aligned LDS tiles.
// ============================================================================

typedef _Float16 f16;
typedef __attribute__((ext_vector_type(16))) _Float16 v16h;
typedef __attribute__((ext_vector_type(8)))  _Float16 v8h;
typedef __attribute__((ext_vector_type(4)))  _Float16 v4h;
typedef __attribute__((ext_vector_type(8)))  float    v8f;
typedef __attribute__((ext_vector_type(4)))  float    v4f;

static inline int rup(int x, int m) { return (x + m - 1) / m * m; }

// ---------------------------------------------------------------------------
// WMMA fragment loaders (contiguous-k layouts, wave32)
// ---------------------------------------------------------------------------
// A: lane (0..15 | 16..31) -> row m0+lane%16; its 16 f16 are two contiguous
// 8-element runs at k0 + (lane/16)*16 and +8. Two ds_load_b128.
__device__ inline v16h frag_a(const f16* A, int lda, int m0, int k0) {
  const int lane = threadIdx.x & 31;
  const f16* p = A + (m0 + (lane & 15)) * lda + k0 + ((lane >> 4) << 4);
  union { v16h v; v8h h[2]; } u;
  u.h[0] = *(const v8h*)p;
  u.h[1] = *(const v8h*)(p + 8);
  return u.v;
}

// B: weights pre-swizzled by pad_w_kernel into fragment order:
// tile (kt,nt) -> 32 lanes x 16 contiguous f16. One 32B load per lane.
__device__ inline v16h frag_b(const f16* B, int Ntiles, int kt, int nt) {
  const int lane = threadIdx.x & 31;
  return *(const v16h*)(B + ((((kt * Ntiles + nt) << 5) + lane) << 4));
}

// C[Mt*16 x Nt*16] = A(lds f16) @ B(global swizzled f16) + bias; tiles split
// round-robin over the block's waves. All control flow is wave-uniform so
// EXEC is all-ones at every v_wmma. B is assumed to span exactly Nt tiles.
template <bool F16OUT>
__device__ inline void gemm(const f16* A, int lda, const f16* B,
                            void* C, int ldc, int Mt, int Nt, int Kp,
                            const float* bias, int Nreal) {
  const int wave = threadIdx.x >> 5;
  const int nw   = blockDim.x >> 5;
  const int lane = threadIdx.x & 31;
  for (int t = wave; t < Mt * Nt; t += nw) {
    const int mt = t / Nt, nt = t - mt * Nt;
    v8f acc = {};
    for (int k0 = 0; k0 < Kp; k0 += 32) {
      v16h a = frag_a(A, lda, mt * 16, k0);
      v16h b = frag_b(B, Nt, k0 >> 5, nt);
      acc = __builtin_amdgcn_wmma_f32_16x16x32_f16(false, a, false, b,
                                                   (short)0, acc, false, false);
    }
    const int n  = nt * 16 + (lane & 15);
    const int mb = mt * 16 + ((lane >> 4) << 3);
    if (bias) {
      const float bv = (n < Nreal) ? bias[n] : 0.0f;
#pragma unroll
      for (int r = 0; r < 8; ++r) acc[r] += bv;
    }
    if (F16OUT) {
      f16* Cp = (f16*)C;
#pragma unroll
      for (int r = 0; r < 8; ++r) Cp[(mb + r) * ldc + n] = (f16)acc[r];
    } else {
      float* Cp = (float*)C;
#pragma unroll
      for (int r = 0; r < 8; ++r) Cp[(mb + r) * ldc + n] = acc[r];
    }
  }
}

// ---------------------------------------------------------------------------
// GVP weight bundle (all matrices pre-swizzled f16; biases f32 from inputs)
// ---------------------------------------------------------------------------
struct GvpW {
  const f16 *wh, *ws, *wv, *wsv;
  const float *ws_b, *wsv_b;
  int vi, kvp;        // input vector channels; padded-32
  int h, hp, kwp;     // hidden vec dim; padded-16 (N of vh gemm); padded-32 (K of wv gemm)
  int si, ksp;        // scalar-in; (si+h) padded-32 (K of s gemm)
  int so, sop, ksvp;  // scalar-out; padded-16; padded-32 (K of gate gemm)
  int vo;             // output vector channels (0, 16 or 32)
};

// One GVP on a 16-row tile. Inputs: scat[e][0..si) f16 (rest of K-span zero),
// va[(d*16+e)][c] f16 (vector inputs, pads zero). Outputs: sout f16 (relu'd if
// requested), vout f32 [(d*16+e)][c] gated. sout cols >= so are zero.
__device__ void gvp_step(const GvpW w, f16* scat, f16* sout, int sld,
                         f16* va, f16* vh, float* vout, float* gate,
                         int relu_act) {
  const int tid = threadIdx.x;
  const int nt  = blockDim.x;
  for (int i = tid; i < 16 * sld; i += nt) sout[i] = (f16)0;
  if (w.vi > 0) {
    for (int i = tid; i < 48 * 64; i += nt) vh[i] = (f16)0;
    __syncthreads();
    // vh = v^T @ wh  (M = 3 dims x 16 rows, K = vi, N = h)
    gemm<true>(va, 64, w.wh, vh, 64, 3, w.hp / 16, w.kvp, nullptr, 0);
    __syncthreads();
    // vn = ||vh||_dims  -> scat[e][si + c]
    for (int i = tid; i < 16 * w.h; i += nt) {
      const int e = i / w.h, c = i - e * w.h;
      const float a0 = (float)vh[e * 64 + c];
      const float a1 = (float)vh[(16 + e) * 64 + c];
      const float a2 = (float)vh[(32 + e) * 64 + c];
      scat[e * sld + w.si + c] = (f16)sqrtf(fmaxf(a0 * a0 + a1 * a1 + a2 * a2, 1e-8f));
    }
  }
  __syncthreads();
  // s_out = [s ; vn] @ ws + b
  gemm<true>(scat, sld, w.ws, sout, sld, 1, w.sop / 16, w.ksp, w.ws_b, w.so);
  __syncthreads();
  if (w.vo > 0) {
    // v_out = vh @ wv ; gate = sigmoid(s_out @ wsv + b)
    gemm<false>(vh, 64, w.wv, vout, 32, 3, w.vo / 16, w.kwp, nullptr, 0);
    gemm<false>(sout, sld, w.wsv, gate, 32, 1, w.vo / 16, w.ksvp, w.wsv_b, w.vo);
    __syncthreads();
    for (int i = tid; i < 16 * w.vo; i += nt) {
      const int e = i / w.vo, c = i - e * w.vo;
      gate[e * 32 + c] = 1.0f / (1.0f + __expf(-gate[e * 32 + c]));
    }
    __syncthreads();
    for (int i = tid; i < 48 * w.vo; i += nt) {
      const int r = i / w.vo, c = i - r * w.vo;
      vout[r * 32 + c] *= gate[(r & 15) * 32 + c];
    }
  }
  if (relu_act) {
    __syncthreads();
    for (int i = tid; i < 16 * w.so; i += nt) {
      const int e = i / w.so, c = i - e * w.so;
      const f16 v = sout[e * sld + c];
      sout[e * sld + c] = ((float)v > 0.0f) ? v : (f16)0;
    }
  }
  __syncthreads();
}

// ---------------------------------------------------------------------------
// Weight pad/convert/swizzle: f32 [K x N] -> f16 fragment-order [Kp/32][Np/16]
// tiles of 32 lanes x 16 f16. Element (k,n): lane=(k%32/16)*16 + n%16,
// slot j = k%16 (contiguous k, as frag_a: b[j] holds K = kg*16 + j for column n).
// ---------------------------------------------------------------------------
__global__ void pad_w_kernel(const float* __restrict__ src, f16* __restrict__ dst,
                             int K, int N, int Kp, int Np) {
  const int i = blockIdx.x * blockDim.x + threadIdx.x;
  if (i >= Kp * Np) return;
  const int k = i / Np, n = i - k * Np;
  const f16 val = (k < K && n < N) ? (f16)src[k * N + n] : (f16)0;
  const int kt = k >> 5, kk = k & 31;
  const int nt = n >> 4, nn = n & 15;
  const int Ntiles = Np >> 4;
  const int lane = ((kk >> 4) << 4) + nn;
  const int j = kk & 15;
  dst[((((kt * Ntiles + nt) << 5) + lane) << 4) + j] = val;
}

// ---------------------------------------------------------------------------
// Node embedding: embed lookup -> LN(9) -> [9]@[9x100]+b
// ---------------------------------------------------------------------------
__global__ void __launch_bounds__(128) node_embed_kernel(
    const int* __restrict__ atoms, const float* __restrict__ embed,
    const float* __restrict__ lng, const float* __restrict__ lnb,
    const float* __restrict__ wsw, const float* __restrict__ wsb,
    float* __restrict__ hs) {
  __shared__ float x[9];
  const int n = blockIdx.x, tid = threadIdx.x;
  if (tid == 0) {
    const int a = atoms[n];
    float t[9], s = 0.f, s2 = 0.f;
    for (int i = 0; i < 9; ++i) { t[i] = embed[a * 9 + i]; s += t[i]; s2 += t[i] * t[i]; }
    const float mu  = s / 9.f;
    const float var = s2 / 9.f - mu * mu;
    const float inv = rsqrtf(var + 1e-5f);
    for (int i = 0; i < 9; ++i) x[i] = (t[i] - mu) * inv * lng[i] + lnb[i];
  }
  __syncthreads();
  if (tid < 100) {
    float a = wsb[tid];
    for (int i = 0; i < 9; ++i) a += x[i] * wsw[i * 100 + tid];
    hs[n * 100 + tid] = a;
  }
}

// ---------------------------------------------------------------------------
// Edge embedding: ln_tuple(16,1) -> GVP (16,1)->(32,1); tiny, one thread/edge
// ---------------------------------------------------------------------------
__global__ void edge_embed_kernel(
    const float* __restrict__ edge_s, const float* __restrict__ edge_v,
    const float* lng, const float* lnb, const float* wh,
    const float* wsw, const float* wsb, const float* wv,
    const float* wsvw, const float* wsvb,
    float* __restrict__ esO, float* __restrict__ evO, int E) {
  const int e = blockIdx.x * blockDim.x + threadIdx.x;
  if (e >= E) return;
  float s[16], mu = 0.f;
  for (int i = 0; i < 16; ++i) { s[i] = edge_s[e * 16 + i]; mu += s[i]; }
  mu *= (1.f / 16.f);
  float var = 0.f;
  for (int i = 0; i < 16; ++i) { const float d = s[i] - mu; var += d * d; }
  var *= (1.f / 16.f);
  const float inv = rsqrtf(var + 1e-5f);
  for (int i = 0; i < 16; ++i) s[i] = (s[i] - mu) * inv * lng[i] + lnb[i];
  float v0 = edge_v[e * 3], v1 = edge_v[e * 3 + 1], v2 = edge_v[e * 3 + 2];
  const float vninv = rsqrtf(fmaxf(v0 * v0 + v1 * v1 + v2 * v2, 1e-8f));
  v0 *= vninv; v1 *= vninv; v2 *= vninv;
  const float whc = wh[0];
  const float h0 = v0 * whc, h1 = v1 * whc, h2 = v2 * whc;
  const float hn = sqrtf(fmaxf(h0 * h0 + h1 * h1 + h2 * h2, 1e-8f));
  float so[32];
  for (int o = 0; o < 32; ++o) {
    float a = wsb[o] + hn * wsw[16 * 32 + o];
    for (int i = 0; i < 16; ++i) a += s[i] * wsw[i * 32 + o];
    so[o] = a;
  }
  float gl = wsvb[0];
  for (int o = 0; o < 32; ++o) gl += so[o] * wsvw[o];
  const float gate = 1.f / (1.f + __expf(-gl));
  const float wvc = wv[0];
  evO[e * 3]     = h0 * wvc * gate;
  evO[e * 3 + 1] = h1 * wvc * gate;
  evO[e * 3 + 2] = h2 * wvc * gate;
  for (int o = 0; o < 32; ++o) esO[e * 32 + o] = so[o];
}

// ---------------------------------------------------------------------------
// Degree count / inverse (for mean aggregation)
// ---------------------------------------------------------------------------
__global__ void cnt_kernel(const int* __restrict__ eidx, int E, float* cnt) {
  const int i = blockIdx.x * blockDim.x + threadIdx.x;
  if (i < E) atomicAdd(&cnt[eidx[E + i]], 1.0f);
}
__global__ void invcnt_kernel(float* cnt, int n) {
  const int i = blockIdx.x * blockDim.x + threadIdx.x;
  if (i < n) cnt[i] = 1.0f / fmaxf(cnt[i], 1.0f);
}

// ---------------------------------------------------------------------------
// Message kernel: one block = 16 edges. Gather -> 3 chained GVPs -> atomic agg
// ---------------------------------------------------------------------------
__global__ void __launch_bounds__(128) msg_kernel(
    const float* __restrict__ hs, const float* __restrict__ hv,
    const float* __restrict__ esB, const float* __restrict__ evB,
    const int* __restrict__ eidx, int E,
    GvpW w0, GvpW w1, GvpW w2,
    float* __restrict__ dsb, float* __restrict__ dvb) {
  __shared__ __align__(32) f16 scat[16 * 288];
  __shared__ __align__(32) f16 sout[16 * 288];
  __shared__ __align__(32) f16 va[48 * 64];
  __shared__ __align__(32) f16 vh[48 * 64];
  __shared__ float vout[48 * 32];
  __shared__ float gate[16 * 32];
  __shared__ int   srcn[16], dstn[16];
  const int tid = threadIdx.x;
  const int e0  = blockIdx.x * 16;
  if (tid < 16) { srcn[tid] = eidx[e0 + tid]; dstn[tid] = eidx[E + e0 + tid]; }
  for (int i = tid; i < 16 * 288; i += 128) scat[i] = (f16)0;
  for (int i = tid; i < 48 * 64; i += 128) va[i] = (f16)0;
  __syncthreads();
  // s_cat = [s[src](100) | es(32) | s[dst](100)] : float4 gather, f16x4 store
  for (int i = tid; i < 16 * 58; i += 128) {
    const int e = i / 58, q = i - e * 58;
    v4f x;
    int c;
    if (q < 25)      { x = *(const v4f*)(hs + srcn[e] * 100 + q * 4);        c = q * 4; }
    else if (q < 33) { x = *(const v4f*)(esB + (e0 + e) * 32 + (q - 25) * 4); c = 100 + (q - 25) * 4; }
    else             { x = *(const v4f*)(hs + dstn[e] * 100 + (q - 33) * 4); c = 132 + (q - 33) * 4; }
    v4h y;
#pragma unroll
    for (int j = 0; j < 4; ++j) y[j] = (f16)x[j];
    *(v4h*)(scat + e * 288 + c) = y;
  }
  // v_cat rows (d*16+e), channels [v[src](16) | ev(1) | v[dst](16)]
  for (int i = tid; i < 48 * 33; i += 128) {
    const int r = i / 33, c = i - r * 33;
    const int d = r >> 4, e = r & 15;
    float x;
    if (c < 16)       x = hv[(srcn[e] * 16 + c) * 3 + d];
    else if (c == 16) x = evB[(e0 + e) * 3 + d];
    else              x = hv[(dstn[e] * 16 + (c - 17)) * 3 + d];
    va[r * 64 + c] = (f16)x;
  }
  __syncthreads();
  gvp_step(w0, scat, sout, 288, va, vh, vout, gate, 1);
  for (int i = tid; i < 16 * 128; i += 128) {
    const int e = i >> 7, c = i & 127;
    scat[e * 288 + c] = (c < 100) ? sout[e * 288 + c] : (f16)0;
  }
  for (int i = tid; i < 48 * 32; i += 128) {
    const int r = i >> 5, c = i & 31;
    va[r * 64 + c] = (c < 16) ? (f16)vout[r * 32 + c] : (f16)0;
  }
  __syncthreads();
  gvp_step(w1, scat, sout, 288, va, vh, vout, gate, 1);
  for (int i = tid; i < 16 * 128; i += 128) {
    const int e = i >> 7, c = i & 127;
    scat[e * 288 + c] = (c < 100) ? sout[e * 288 + c] : (f16)0;
  }
  for (int i = tid; i < 48 * 32; i += 128) {
    const int r = i >> 5, c = i & 31;
    va[r * 64 + c] = (c < 16) ? (f16)vout[r * 32 + c] : (f16)0;
  }
  __syncthreads();
  gvp_step(w2, scat, sout, 288, va, vh, vout, gate, 0);
  // mean-aggregate (sum here; divide by count in residual kernel)
  for (int i = tid; i < 16 * 100; i += 128) {
    const int e = i / 100, c = i - e * 100;
    atomicAdd(&dsb[dstn[e] * 100 + c], (float)sout[e * 288 + c]);
  }
  for (int i = tid; i < 16 * 48; i += 128) {
    const int e = i / 48, j = i - e * 48;
    const int c = j / 3, d = j - c * 3;
    atomicAdd(&dvb[(dstn[e] * 16 + c) * 3 + d], vout[(d * 16 + e) * 32 + c]);
  }
}

// ---------------------------------------------------------------------------
// Residual + ln_tuple: hs = LN(hs + ds*scale), hv = (hv + dv*scale)/vrms
// ---------------------------------------------------------------------------
__global__ void __launch_bounds__(128) residual_ln_kernel(
    float* __restrict__ hs, float* __restrict__ hv,
    const float* __restrict__ dsb, const float* __restrict__ dvb,
    const float* __restrict__ inv_cnt,
    const float* __restrict__ g, const float* __restrict__ b) {
  __shared__ float red[128], red2[128], sval[100], vbuf[48], vsh[1];
  const int n = blockIdx.x, tid = threadIdx.x;
  const float sc = inv_cnt ? inv_cnt[n] : 1.0f;
  float x = 0.f;
  if (tid < 100) { x = hs[n * 100 + tid] + dsb[n * 100 + tid] * sc; sval[tid] = x; }
  red[tid]  = (tid < 100) ? x : 0.f;
  red2[tid] = (tid < 100) ? x * x : 0.f;
  if (tid < 48) vbuf[tid] = hv[n * 48 + tid] + dvb[n * 48 + tid] * sc;
  __syncthreads();
  for (int s = 64; s > 0; s >>= 1) {
    if (tid < s) { red[tid] += red[tid + s]; red2[tid] += red2[tid + s]; }
    __syncthreads();
  }
  const float mu  = red[0] * 0.01f;
  const float var = red2[0] * 0.01f - mu * mu;
  const float inv = rsqrtf(var + 1e-5f);
  __syncthreads();
  if (tid < 16) {
    float a = 0.f;
    for (int d = 0; d < 3; ++d) { const float v = vbuf[tid * 3 + d]; a += v * v; }
    red[tid] = fmaxf(a, 1e-8f);
  }
  __syncthreads();
  if (tid == 0) {
    float s = 0.f;
    for (int c = 0; c < 16; ++c) s += red[c];
    vsh[0] = rsqrtf(s * (1.f / 16.f));
  }
  __syncthreads();
  if (tid < 100) hs[n * 100 + tid] = (sval[tid] - mu) * inv * g[tid] + b[tid];
  if (tid < 48)  hv[n * 48 + tid]  = vbuf[tid] * vsh[0];
}

// ---------------------------------------------------------------------------
// Feed-forward kernel: one block = 16 nodes, GVP(100,16->400,32,relu) then
// GVP(400,32->100,16)
// ---------------------------------------------------------------------------
__global__ void __launch_bounds__(128) ff_kernel(
    const float* __restrict__ hs, const float* __restrict__ hv,
    GvpW w0, GvpW w1, float* __restrict__ fsb, float* __restrict__ fvb) {
  __shared__ __align__(32) f16 scat[16 * 448];
  __shared__ __align__(32) f16 sout[16 * 448];
  __shared__ __align__(32) f16 va[48 * 64];
  __shared__ __align__(32) f16 vh[48 * 64];
  __shared__ float vout[48 * 32];
  __shared__ float gate[16 * 32];
  const int tid = threadIdx.x;
  const int n0  = blockIdx.x * 16;
  for (int i = tid; i < 16 * 448; i += 128) scat[i] = (f16)0;
  for (int i = tid; i < 48 * 64; i += 128) va[i] = (f16)0;
  __syncthreads();
  for (int i = tid; i < 16 * 25; i += 128) {
    const int e = i / 25, q = i - e * 25;
    const v4f x = *(const v4f*)(hs + (n0 + e) * 100 + q * 4);
    v4h y;
#pragma unroll
    for (int j = 0; j < 4; ++j) y[j] = (f16)x[j];
    *(v4h*)(scat + e * 448 + q * 4) = y;
  }
  for (int i = tid; i < 48 * 16; i += 128) {
    const int r = i / 16, c = i - r * 16;
    const int d = r >> 4, e = r & 15;
    va[r * 64 + c] = (f16)hv[((n0 + e) * 16 + c) * 3 + d];
  }
  __syncthreads();
  gvp_step(w0, scat, sout, 448, va, vh, vout, gate, 1);
  for (int i = tid; i < 16 * 448; i += 128) {
    const int e = i / 448, c = i - e * 448;
    scat[e * 448 + c] = (c < 400) ? sout[e * 448 + c] : (f16)0;
  }
  for (int i = tid; i < 48 * 32; i += 128) {
    const int r = i >> 5, c = i & 31;
    va[r * 64 + c] = (f16)vout[r * 32 + c];
  }
  __syncthreads();
  gvp_step(w1, scat, sout, 448, va, vh, vout, gate, 0);
  for (int i = tid; i < 16 * 25; i += 128) {
    const int e = i / 25, q = i - e * 25;
    const v4h y = *(const v4h*)(sout + e * 448 + q * 4);
    v4f x;
#pragma unroll
    for (int j = 0; j < 4; ++j) x[j] = (float)y[j];
    *(v4f*)(fsb + (n0 + e) * 100 + q * 4) = x;
  }
  for (int i = tid; i < 16 * 48; i += 128) {
    const int e = i / 48, j = i - e * 48;
    const int c = j / 3, d = j - c * 3;
    fvb[((n0 + e) * 16 + c) * 3 + d] = vout[(d * 16 + e) * 32 + c];
  }
}

// ---------------------------------------------------------------------------
// Output head: ln_tuple -> Wout GVP (relu) -> dense1(relu) -> dense2
// ---------------------------------------------------------------------------
__global__ void __launch_bounds__(128) out_kernel(
    const float* __restrict__ hs, const float* __restrict__ hv,
    const float* lng, const float* lnb, GvpW w,
    const f16* __restrict__ w1p, const float* b1,
    const f16* __restrict__ w2p, const float* b2,
    float* __restrict__ nodeout) {
  __shared__ __align__(32) f16 scat[16 * 288];
  __shared__ __align__(32) f16 sout[16 * 288];
  __shared__ __align__(32) f16 va[48 * 64];
  __shared__ __align__(32) f16 vh[48 * 64];
  __shared__ float out2[16 * 32];
  __shared__ float mu_s[16], inv_s[16], vr_s[16];
  const int tid = threadIdx.x;
  const int n0  = blockIdx.x * 16;
  if (tid < 16) {
    const int n = n0 + tid;
    float s = 0.f, s2 = 0.f;
    for (int c = 0; c < 100; ++c) { const float x = hs[n * 100 + c]; s += x; s2 += x * x; }
    const float mu = s * 0.01f;
    mu_s[tid]  = mu;
    inv_s[tid] = rsqrtf(s2 * 0.01f - mu * mu + 1e-5f);
    float vs = 0.f;
    for (int c = 0; c < 16; ++c) {
      float a = 0.f;
      for (int d = 0; d < 3; ++d) { const float v = hv[(n * 16 + c) * 3 + d]; a += v * v; }
      vs += fmaxf(a, 1e-8f);
    }
    vr_s[tid] = rsqrtf(vs * (1.f / 16.f));
  }
  for (int i = tid; i < 16 * 288; i += 128) scat[i] = (f16)0;
  for (int i = tid; i < 48 * 64; i += 128) va[i] = (f16)0;
  __syncthreads();
  for (int i = tid; i < 16 * 100; i += 128) {
    const int e = i / 100, c = i - e * 100;
    scat[e * 288 + c] =
        (f16)((hs[(n0 + e) * 100 + c] - mu_s[e]) * inv_s[e] * lng[c] + lnb[c]);
  }
  for (int i = tid; i < 48 * 16; i += 128) {
    const int r = i / 16, c = i - r * 16;
    const int d = r >> 4, e = r & 15;
    va[r * 64 + c] = (f16)(hv[((n0 + e) * 16 + c) * 3 + d] * vr_s[e]);
  }
  __syncthreads();
  gvp_step(w, scat, sout, 288, va, vh, nullptr, nullptr, 1);  // vo==0
  // dense1: [16x128]@[128x208] + b1, relu   (output into scat)
  for (int i = tid; i < 16 * 288; i += 128) scat[i] = (f16)0;
  __syncthreads();
  gemm<true>(sout, 288, w1p, scat, 288, 1, 13, 128, b1, 200);
  __syncthreads();
  for (int i = tid; i < 16 * 200; i += 128) {
    const int e = i / 200, c = i - e * 200;
    const f16 v = scat[e * 288 + c];
    scat[e * 288 + c] = ((float)v > 0.0f) ? v : (f16)0;
  }
  __syncthreads();
  // dense2: [16x224]@[224x32] + b2
  gemm<false>(scat, 288, w2p, out2, 32, 1, 2, 224, b2, 20);
  __syncthreads();
  for (int i = tid; i < 16 * 20; i += 128) {
    const int e = i / 20, c = i - e * 20;
    nodeout[(n0 + e) * 20 + c] = out2[e * 32 + c];
  }
}

__global__ void out_gather_kernel(const float* __restrict__ nodeout,
                                  const int* __restrict__ ca,
                                  const int* __restrict__ ptr,
                                  float* __restrict__ out, int NG) {
  const int i = blockIdx.x * blockDim.x + threadIdx.x;
  if (i >= NG * 20) return;
  const int b = i / 20, c = i - b * 20;
  out[i] = nodeout[(ca[b] + ptr[b]) * 20 + c];
}

// ===========================================================================
// Host orchestration
// ===========================================================================
// The harness flattens the setup_inputs() pytree. Two plausible orders:
//   sorted    : jax tree_flatten (dict keys sorted alphabetically)
//   insertion : raw dict insertion order
// Distinguish at runtime: in_sizes[1]==32 -> sorted (ca_idx), else insertion.
struct LeafMap {
  bool srt;
  int atoms, ca, eidx, es, ev, ptr, embed;
  int wv_ln_g, wv_ln_b, wv_ws_w, wv_ws_b;
  int we_ln_g, we_ln_b, we_wh, we_ws_w, we_ws_b, we_wv, we_wsv_w, we_wsv_b;
  int wout_ln_g, wout_ln_b, wout_wh, wout_ws_w, wout_ws_b;
  int d_w1, d_b1, d_w2, d_b2;
  // gvp g: 0..4 = msg0,msg1,msg2,ff0,ff1; role r: 0 wh,1 ws_w,2 ws_b,3 wv,4 wsv_w,5 wsv_b
  int gvp(int l, int g, int r) const {
    static const int SG[5] = {16, 22, 28, 0, 6};
    static const int SR[6] = {0, 2, 1, 5, 4, 3};
    static const int IG[5] = {0, 6, 12, 18, 24};
    return srt ? 27 + l * 34 + SG[g] + SR[r] : 19 + l * 34 + IG[g] + r;
  }
  int ln(int l, int which, int gb) const {  // gb: 0=g, 1=b
    return srt ? 27 + l * 34 + 12 + which * 2 + (1 - gb)
               : 19 + l * 34 + 30 + which * 2 + gb;
  }
};

extern "C" void kernel_launch(void* const* d_in, const int* in_sizes, int n_in,
                              void* d_out, int out_size, void* d_ws, size_t ws_size,
                              hipStream_t stream) {
  (void)out_size; (void)ws_size;
  if (n_in < 198) return;
  const int N = 6400, E = 102400, NG = 32;

  LeafMap M{};
  M.srt = (in_sizes[1] == 32);
  if (M.srt) {
    M.atoms = 0; M.ca = 1; M.eidx = 2; M.es = 3; M.ev = 4;
    M.we_wh = 5; M.we_ws_b = 6; M.we_ws_w = 7; M.we_wsv_b = 8; M.we_wsv_w = 9;
    M.we_wv = 10; M.we_ln_b = 11; M.we_ln_g = 12;
    M.wout_wh = 13; M.wout_ws_b = 14; M.wout_ws_w = 15; M.wout_ln_b = 16; M.wout_ln_g = 17;
    M.wv_ws_b = 18; M.wv_ws_w = 19; M.wv_ln_b = 20; M.wv_ln_g = 21;
    M.d_b1 = 22; M.d_b2 = 23; M.d_w1 = 24; M.d_w2 = 25; M.embed = 26; M.ptr = 197;
  } else {
    M.atoms = 0; M.eidx = 1; M.es = 2; M.ev = 3; M.ca = 4; M.ptr = 5; M.embed = 6;
    M.wv_ln_g = 7; M.wv_ln_b = 8; M.wv_ws_w = 9; M.wv_ws_b = 10;
    M.we_ln_g = 11; M.we_ln_b = 12; M.we_wh = 13; M.we_ws_w = 14; M.we_ws_b = 15;
    M.we_wv = 16; M.we_wsv_w = 17; M.we_wsv_b = 18;
    M.wout_ln_g = 189; M.wout_ln_b = 190; M.wout_wh = 191; M.wout_ws_w = 192; M.wout_ws_b = 193;
    M.d_w1 = 194; M.d_b1 = 195; M.d_w2 = 196; M.d_b2 = 197;
  }

  auto F = [&](int i) -> const float* { return (const float*)d_in[i]; };
  auto I = [&](int i) -> const int*   { return (const int*)d_in[i]; };

  // --- workspace bump allocator ---
  size_t off = 0;
  auto alloc = [&](size_t bytes) -> void* {
    void* p = (char*)d_ws + off;
    off += (bytes + 255) & ~(size_t)255;
    return p;
  };
  float* hs      = (float*)alloc((size_t)N * 100 * 4);
  float* hv      = (float*)alloc((size_t)N * 48 * 4);
  float* esB     = (float*)alloc((size_t)E * 32 * 4);
  float* evB     = (float*)alloc((size_t)E * 3 * 4);
  float* dsb     = (float*)alloc((size_t)N * 100 * 4);
  float* dvb     = (float*)alloc((size_t)N * 48 * 4);
  float* fsb     = (float*)alloc((size_t)N * 400 * 4);
  float* fvb     = (float*)alloc((size_t)N * 48 * 4);
  float* cnt     = (float*)alloc((size_t)N * 4);
  float* nodeout = (float*)alloc((size_t)N * 20 * 4);

  auto prepw = [&](const float* src, int K, int Ncol, int Kp, int Np) -> f16* {
    f16* dst = (f16*)alloc((size_t)Kp * Np * sizeof(f16));
    const int total = Kp * Np;
    pad_w_kernel<<<(total + 255) / 256, 256, 0, stream>>>(src, dst, K, Ncol, Kp, Np);
    return dst;
  };
  auto mkgvp = [&](const float* wh_, const float* wsw_, const float* wsb_,
                   const float* wv_, const float* wsvw_, const float* wsvb_,
                   int vi, int vo, int si, int so) -> GvpW {
    GvpW g{};
    g.vi = vi; g.vo = vo; g.si = si; g.so = so;
    const int h = (vi > vo) ? vi : vo;
    g.h = h; g.kvp = rup(vi, 32); g.hp = rup(h, 16); g.kwp = rup(h, 32);
    g.ksp = rup(si + h, 32); g.sop = rup(so, 16); g.ksvp = rup(so, 32);
    g.ws_b = wsb_; g.wsv_b = wsvb_;
    g.wh = prepw(wh_, vi, h, g.kvp, g.hp);
    g.ws = prepw(wsw_, si + h, so, g.ksp, g.sop);
    if (vo > 0) {
      g.wv  = prepw(wv_, h, vo, g.kwp, vo);
      g.wsv = prepw(wsvw_, so, vo, g.ksvp, vo);
    }
    return g;
  };

  // --- pre-swizzle all WMMA weights (f32 -> fragment-order f16, hot in L2) ---
  GvpW msgw[5][3], ffw[5][2];
  const float *ln0g[5], *ln0b[5], *ln1g[5], *ln1b[5];
  for (int l = 0; l < 5; ++l) {
    msgw[l][0] = mkgvp(F(M.gvp(l,0,0)), F(M.gvp(l,0,1)), F(M.gvp(l,0,2)),
                       F(M.gvp(l,0,3)), F(M.gvp(l,0,4)), F(M.gvp(l,0,5)), 33, 16, 232, 100);
    msgw[l][1] = mkgvp(F(M.gvp(l,1,0)), F(M.gvp(l,1,1)), F(M.gvp(l,1,2)),
                       F(M.gvp(l,1,3)), F(M.gvp(l,1,4)), F(M.gvp(l,1,5)), 16, 16, 100, 100);
    msgw[l][2] = mkgvp(F(M.gvp(l,2,0)), F(M.gvp(l,2,1)), F(M.gvp(l,2,2)),
                       F(M.gvp(l,2,3)), F(M.gvp(l,2,4)), F(M.gvp(l,2,5)), 16, 16, 100, 100);
    ffw[l][0]  = mkgvp(F(M.gvp(l,3,0)), F(M.gvp(l,3,1)), F(M.gvp(l,3,2)),
                       F(M.gvp(l,3,3)), F(M.gvp(l,3,4)), F(M.gvp(l,3,5)), 16, 32, 100, 400);
    ffw[l][1]  = mkgvp(F(M.gvp(l,4,0)), F(M.gvp(l,4,1)), F(M.gvp(l,4,2)),
                       F(M.gvp(l,4,3)), F(M.gvp(l,4,4)), F(M.gvp(l,4,5)), 32, 16, 400, 100);
    ln0g[l] = F(M.ln(l,0,0)); ln0b[l] = F(M.ln(l,0,1));
    ln1g[l] = F(M.ln(l,1,0)); ln1b[l] = F(M.ln(l,1,1));
  }
  GvpW woutw = mkgvp(F(M.wout_wh), F(M.wout_ws_w), F(M.wout_ws_b),
                     nullptr, nullptr, nullptr, 16, 0, 100, 100);
  f16* w1p = prepw(F(M.d_w1), 100, 200, 128, 208);
  f16* w2p = prepw(F(M.d_w2), 200, 20, 224, 32);

  // --- embeddings ---
  node_embed_kernel<<<N, 128, 0, stream>>>(I(M.atoms), F(M.embed),
      F(M.wv_ln_g), F(M.wv_ln_b), F(M.wv_ws_w), F(M.wv_ws_b), hs);
  (void)hipMemsetAsync(hv, 0, (size_t)N * 48 * 4, stream);
  edge_embed_kernel<<<(E + 255) / 256, 256, 0, stream>>>(F(M.es), F(M.ev),
      F(M.we_ln_g), F(M.we_ln_b), F(M.we_wh), F(M.we_ws_w), F(M.we_ws_b),
      F(M.we_wv), F(M.we_wsv_w), F(M.we_wsv_b), esB, evB, E);
  (void)hipMemsetAsync(cnt, 0, (size_t)N * 4, stream);
  cnt_kernel<<<(E + 255) / 256, 256, 0, stream>>>(I(M.eidx), E, cnt);
  invcnt_kernel<<<(N + 255) / 256, 256, 0, stream>>>(cnt, N);

  // --- 5 GNN layers ---
  for (int l = 0; l < 5; ++l) {
    (void)hipMemsetAsync(dsb, 0, (size_t)N * 100 * 4, stream);
    (void)hipMemsetAsync(dvb, 0, (size_t)N * 48 * 4, stream);
    msg_kernel<<<E / 16, 128, 0, stream>>>(hs, hv, esB, evB, I(M.eidx), E,
                                           msgw[l][0], msgw[l][1], msgw[l][2], dsb, dvb);
    residual_ln_kernel<<<N, 128, 0, stream>>>(hs, hv, dsb, dvb, cnt, ln0g[l], ln0b[l]);
    ff_kernel<<<N / 16, 128, 0, stream>>>(hs, hv, ffw[l][0], ffw[l][1], fsb, fvb);
    residual_ln_kernel<<<N, 128, 0, stream>>>(hs, hv, fsb, fvb, nullptr, ln1g[l], ln1b[l]);
  }

  // --- output head + gather ---
  out_kernel<<<N / 16, 128, 0, stream>>>(hs, hv, F(M.wout_ln_g), F(M.wout_ln_b),
                                         woutw, w1p, F(M.d_b1), w2p, F(M.d_b2), nodeout);
  out_gather_kernel<<<(NG * 20 + 127) / 128, 128, 0, stream>>>(
      nodeout, I(M.ca), I(M.ptr), (float*)d_out, NG);
}
